// DynamicFFNLayer_1039382085962
// MI455X (gfx1250) — matrix operand order repaired and checked by reading, not compiled, not run on hardware
//
#include <hip/hip_runtime.h>
#include <hip/hip_bf16.h>

#define D_MODEL 1024
#define D_FF    4096
#define NTOK    8192   // BATCH*SEQ = 4*2048
#define TOPK_DEF 1024

typedef __bf16 bf16;
typedef __attribute__((ext_vector_type(16))) __bf16 v16bf;
typedef __attribute__((ext_vector_type(8)))  __bf16 v8bf;
typedef __attribute__((ext_vector_type(8)))  float  v8f;

// ---------------------------------------------------------------- utilities

__device__ __forceinline__ float gelu_exact(float x) {
    return 0.5f * x * (1.0f + erff(x * 0.70710678118654752440f));
}

// Order-preserving 16-bit key for bf16 (monotone: a<b  <=>  key(a)<key(b))
__device__ __forceinline__ unsigned short okey_bf16(bf16 x) {
    union { bf16 b; unsigned short u; } cv; cv.b = x;
    unsigned short u = cv.u;
    return (u & 0x8000u) ? (unsigned short)(~u) : (unsigned short)(u | 0x8000u);
}

// ---------------------------------------------------------------- f32 -> bf16

__global__ __launch_bounds__(256)
void cast_f32_bf16(const float* __restrict__ in, bf16* __restrict__ out, int n) {
    int i = blockIdx.x * 256 + threadIdx.x;
    if (i < n) out[i] = (bf16)in[i];
}

// -------------------------------------------------- LayerNorm + bf16 cast of x
// One workgroup (256 threads) per token row of 1024.

__global__ __launch_bounds__(256)
void ln_and_cast(const float* __restrict__ x, const float* __restrict__ gamma,
                 const float* __restrict__ beta, bf16* __restrict__ xn,
                 bf16* __restrict__ xb) {
    const int row = blockIdx.x;
    const int tid = threadIdx.x;
    const float* xr = x + (size_t)row * D_MODEL;

    __shared__ float rs[256];
    __shared__ float rq[256];

    float v[4];
    float s = 0.f, q = 0.f;
#pragma unroll
    for (int j = 0; j < 4; ++j) {
        v[j] = xr[tid + j * 256];
        s += v[j];
        q += v[j] * v[j];
    }
    rs[tid] = s; rq[tid] = q;
    __syncthreads();
    for (int step = 128; step > 0; step >>= 1) {
        if (tid < step) { rs[tid] += rs[tid + step]; rq[tid] += rq[tid + step]; }
        __syncthreads();
    }
    const float mean = rs[0] * (1.0f / D_MODEL);
    const float var  = rq[0] * (1.0f / D_MODEL) - mean * mean;
    const float rstd = rsqrtf(var + 1e-5f);

#pragma unroll
    for (int j = 0; j < 4; ++j) {
        int c = tid + j * 256;
        float nv = (v[j] - mean) * rstd * gamma[c] + beta[c];
        xn[(size_t)row * D_MODEL + c] = (bf16)nv;
        xb[(size_t)row * D_MODEL + c] = (bf16)v[j];
    }
}

// -------------------------------------------------- top-k threshold (radix select)
// One workgroup per row of d_ff=4096 bf16 scores; finds k-th largest value's
// order-key via two 8-bit-digit histogram passes. thr[row] = key of k-th largest.

__global__ __launch_bounds__(256)
void topk_threshold(const bf16* __restrict__ S, unsigned short* __restrict__ thr,
                    const int* __restrict__ kptr) {
    const int row = blockIdx.x;
    const int tid = threadIdx.x;
    const unsigned int k = (unsigned int)(kptr ? kptr[0] : TOPK_DEF);
    const bf16* sr = S + (size_t)row * D_FF;

    __shared__ unsigned int hist[256];
    __shared__ unsigned int sel_hi;
    __shared__ unsigned int rem;

    // pass 1: high byte
    hist[tid] = 0u;
    __syncthreads();
    for (int i = tid; i < D_FF; i += 256)
        atomicAdd(&hist[okey_bf16(sr[i]) >> 8], 1u);
    __syncthreads();
    if (tid == 0) {
        unsigned int cum = 0; int b = 255;
        for (; b > 0; --b) {
            if (cum + hist[b] >= k) break;
            cum += hist[b];
        }
        sel_hi = (unsigned int)b;
        rem = k - cum;
    }
    __syncthreads();
    const unsigned int hb = sel_hi;
    const unsigned int r2 = rem;
    __syncthreads();

    // pass 2: low byte among elements whose high byte == hb
    hist[tid] = 0u;
    __syncthreads();
    for (int i = tid; i < D_FF; i += 256) {
        unsigned short key = okey_bf16(sr[i]);
        if ((key >> 8) == hb) atomicAdd(&hist[key & 255u], 1u);
    }
    __syncthreads();
    if (tid == 0) {
        unsigned int cum = 0; int l = 255;
        for (; l > 0; --l) {
            if (cum + hist[l] >= r2) break;
            cum += hist[l];
        }
        thr[row] = (unsigned short)((hb << 8) | (unsigned int)l);
    }
}

// -------------------------------------------------- WMMA GEMM: C = A * B^T
// A: [M,K] bf16 row-major.  B: [N,K] bf16 row-major (so B^T is K x N).
// Block: 256 threads = 8 waves arranged 4 (M) x 2 (N). Block tile 256x128;
// each wave computes 64x64 = 4x4 tiles of 16x16. K-step 32 via
// v_wmma_f32_16x16x32_bf16. Per K-step: 8 fragments (16 b128 loads) feed
// 16 WMMAs -> 2 WMMA per fragment load.
//
// Fragment layouts per CDNA5 ISA 7.12.2 (wave32):
//   A 16x32 bf16: lane holds row M=lane%16; K chunks [kb..kb+7] in v0-3 and
//                 [kb+16..kb+23] in v4-7, kb = (lane/16)*8.
//   B 32x16 bf16: lane holds col N=lane%16; same K chunks as A:
//                 [kb..kb+7] in v0-3, [kb+16..kb+23] in v4-7.
// C/D 16x16 f32: vgpr e -> M = e + (lane/16)*8, N = lane%16.
//
// MODE: 0 = store f32, 1 = store bf16, 2 = gelu -> bf16,
//       3 = topk-mask (S,thr) + gelu -> bf16.

template<int MODE>
__global__ __launch_bounds__(256)
void gemm_bt_wmma(const bf16* __restrict__ A, const bf16* __restrict__ B,
                  float* __restrict__ Cf, bf16* __restrict__ Cb,
                  const bf16* __restrict__ S, const unsigned short* __restrict__ thr,
                  int M, int N, int K) {
    const int lane = threadIdx.x & 31;
    const int wid  = threadIdx.x >> 5;
    const int wm   = wid >> 1;     // 0..3 : wave row in block (64 rows each)
    const int wn   = wid & 1;      // 0..1 : wave col in block (64 cols each)
    const int bm   = blockIdx.y * 256;
    const int bn   = blockIdx.x * 128;

    const int l16 = lane & 15;
    const int lhi = lane >> 4;

    const int rowA0 = bm + wm * 64;
    const int colB0 = bn + wn * 64;

    // Per-lane base pointers including lane-dependent K offsets.
    const bf16* Abase = A + (size_t)(rowA0 + l16) * K + (lhi << 3);   // + {0,8}
    const bf16* Bbase = B + (size_t)(colB0 + l16) * K + (lhi << 3);   // + {0,8}

    v8f acc[4][4];
#pragma unroll
    for (int i = 0; i < 4; ++i)
#pragma unroll
        for (int j = 0; j < 4; ++j)
#pragma unroll
            for (int e = 0; e < 8; ++e)
                acc[i][j][e] = 0.0f;

    for (int k = 0; k < K; k += 32) {
        if (k + 32 < K) {   // hint next K tile into cache (global_prefetch_b8)
            __builtin_prefetch((const void*)(Abase + k + 32), 0, 0);
            __builtin_prefetch((const void*)(Bbase + k + 32), 0, 0);
        }

        v16bf fa[4], fb[4];
#pragma unroll
        for (int i = 0; i < 4; ++i) {
            const bf16* p = Abase + (size_t)(i * 16) * K + k;
            v8bf lo = *(const v8bf*)(p);        // K = kb .. kb+7
            v8bf hi = *(const v8bf*)(p + 16);   // K = kb+16 .. kb+23
            fa[i] = __builtin_shufflevector(lo, hi,
                        0,1,2,3,4,5,6,7,8,9,10,11,12,13,14,15);
        }
#pragma unroll
        for (int j = 0; j < 4; ++j) {
            const bf16* p = Bbase + (size_t)(j * 16) * K + k;
            v8bf lo = *(const v8bf*)(p);        // K = kb .. kb+7
            v8bf hi = *(const v8bf*)(p + 16);   // K = kb+16 .. kb+23
            fb[j] = __builtin_shufflevector(lo, hi,
                        0,1,2,3,4,5,6,7,8,9,10,11,12,13,14,15);
        }

#pragma unroll
        for (int i = 0; i < 4; ++i)
#pragma unroll
            for (int j = 0; j < 4; ++j)
                acc[i][j] = __builtin_amdgcn_wmma_f32_16x16x32_bf16(
                    false, fa[i], false, fb[j], (short)0, acc[i][j], false, false);
    }

    // Epilogue
#pragma unroll
    for (int i = 0; i < 4; ++i) {
#pragma unroll
        for (int j = 0; j < 4; ++j) {
            const int col = bn + wn * 64 + j * 16 + l16;
#pragma unroll
            for (int e = 0; e < 8; ++e) {
                const int row = bm + wm * 64 + i * 16 + lhi * 8 + e;
                const size_t off = (size_t)row * N + col;
                float v = acc[i][j][e];
                if (MODE == 0) {
                    Cf[off] = v;
                } else if (MODE == 1) {
                    Cb[off] = (bf16)v;
                } else if (MODE == 2) {
                    Cb[off] = (bf16)gelu_exact(v);
                } else {
                    unsigned short key = okey_bf16(S[off]);
                    float g = (key >= thr[row]) ? gelu_exact(v) : 0.0f;
                    Cb[off] = (bf16)g;
                }
            }
        }
    }
}

// ---------------------------------------------------------------- launcher

extern "C" void kernel_launch(void* const* d_in, const int* in_sizes, int n_in,
                              void* d_out, int out_size, void* d_ws, size_t ws_size,
                              hipStream_t stream) {
    const float* x    = (const float*)d_in[0];   // [NTOK, D_MODEL]
    const float* W1   = (const float*)d_in[1];   // [D_FF, D_MODEL]
    const float* W2   = (const float*)d_in[2];   // [D_MODEL, D_FF]
    const float* Wr1  = (const float*)d_in[3];   // [D_MODEL, D_MODEL]
    const float* Wr2  = (const float*)d_in[4];   // [D_FF, D_MODEL]
    const float* g    = (const float*)d_in[5];   // [D_MODEL]
    const float* bta  = (const float*)d_in[6];   // [D_MODEL]
    const int*   kptr = (n_in > 7) ? (const int*)d_in[7] : nullptr;

    // workspace layout (bf16 staging buffers)
    char* ws = (char*)d_ws;
    size_t off = 0;
    auto alloc = [&](size_t bytes) -> char* {
        char* p = ws + off;
        off += (bytes + 255) & ~(size_t)255;
        return p;
    };
    bf16* xb   = (bf16*)alloc((size_t)NTOK * D_MODEL * sizeof(bf16));  // x as bf16
    bf16* xn   = (bf16*)alloc((size_t)NTOK * D_MODEL * sizeof(bf16));  // LN(x)
    bf16* w1b  = (bf16*)alloc((size_t)D_FF * D_MODEL * sizeof(bf16));
    bf16* w2b  = (bf16*)alloc((size_t)D_MODEL * D_FF * sizeof(bf16));
    bf16* wr1b = (bf16*)alloc((size_t)D_MODEL * D_MODEL * sizeof(bf16));
    bf16* wr2b = (bf16*)alloc((size_t)D_FF * D_MODEL * sizeof(bf16));
    bf16* h    = (bf16*)alloc((size_t)NTOK * D_MODEL * sizeof(bf16));  // router hidden
    bf16* s    = (bf16*)alloc((size_t)NTOK * D_FF * sizeof(bf16));     // router scores
    bf16* a    = (bf16*)alloc((size_t)NTOK * D_FF * sizeof(bf16));     // masked gelu act
    unsigned short* thr = (unsigned short*)alloc((size_t)NTOK * sizeof(unsigned short));
    (void)ws_size;

    // 1) weight casts
    {
        int n;
        n = D_FF * D_MODEL;    cast_f32_bf16<<<(n + 255) / 256, 256, 0, stream>>>(W1,  w1b,  n);
        n = D_MODEL * D_FF;    cast_f32_bf16<<<(n + 255) / 256, 256, 0, stream>>>(W2,  w2b,  n);
        n = D_MODEL * D_MODEL; cast_f32_bf16<<<(n + 255) / 256, 256, 0, stream>>>(Wr1, wr1b, n);
        n = D_FF * D_MODEL;    cast_f32_bf16<<<(n + 255) / 256, 256, 0, stream>>>(Wr2, wr2b, n);
    }

    // 2) LayerNorm + bf16 cast of x
    ln_and_cast<<<NTOK, 256, 0, stream>>>(x, g, bta, xn, xb);

    // 3) h = gelu(xn @ Wr1^T)   [NTOK, D_MODEL]
    gemm_bt_wmma<2><<<dim3(D_MODEL / 128, NTOK / 256), 256, 0, stream>>>(
        xn, wr1b, nullptr, h, nullptr, nullptr, NTOK, D_MODEL, D_MODEL);

    // 4) s = h @ Wr2^T          [NTOK, D_FF]
    gemm_bt_wmma<1><<<dim3(D_FF / 128, NTOK / 256), 256, 0, stream>>>(
        h, wr2b, nullptr, s, nullptr, nullptr, NTOK, D_FF, D_MODEL);

    // 5) per-row top-k threshold
    topk_threshold<<<NTOK, 256, 0, stream>>>(s, thr, kptr);

    // 6) a = gelu((x @ W1^T) * mask)   [NTOK, D_FF]
    gemm_bt_wmma<3><<<dim3(D_FF / 128, NTOK / 256), 256, 0, stream>>>(
        xb, w1b, nullptr, a, s, thr, NTOK, D_FF, D_MODEL);

    // 7) out = a @ W2^T (fp32)   [NTOK, D_MODEL]
    gemm_bt_wmma<0><<<dim3(D_MODEL / 128, NTOK / 256), 256, 0, stream>>>(
        a, w2b, (float*)d_out, nullptr, nullptr, nullptr, NTOK, D_MODEL, D_FF);

    (void)in_sizes; (void)out_size;
}
